// MultiHeadAttention_2611340116465
// MI455X (gfx1250) — matrix operand order repaired and checked by reading, not compiled, not run on hardware
//
#include <hip/hip_runtime.h>
#include <hip/hip_bf16.h>

// ---------------------------------------------------------------------------
// CDNA5 (gfx1250) MHA forward: bf16 WMMA GEMM projections + flash attention,
// with Tensor Data Mover (TDM) staging of attention K-tiles into LDS.
// ---------------------------------------------------------------------------

typedef __attribute__((ext_vector_type(16))) __bf16 v16bf;
typedef __attribute__((ext_vector_type(8)))  __bf16 v8bf;
typedef __attribute__((ext_vector_type(4)))  __bf16 v4bf;
typedef __attribute__((ext_vector_type(8)))  float  v8f;
typedef __attribute__((ext_vector_type(4)))  unsigned int u32x4;
typedef __attribute__((ext_vector_type(8)))  int  i32x8;
typedef __attribute__((ext_vector_type(4)))  int  i32x4;

static __device__ inline v8f zero8() {
    v8f z;
#pragma unroll
    for (int i = 0; i < 8; ++i) z[i] = 0.0f;
    return z;
}

union FragU {
    struct { v8bf lo, hi; } p;
    v16bf v;
};

static __device__ inline v16bf mkfrag(v8bf lo, v8bf hi) {
    FragU u; u.p.lo = lo; u.p.hi = hi; return u.v;
}

// A-matrix fragment (16x32 bf16, MxK): lane = M row (lane&15), khalf = lane>>4.
// Per ISA: elems 0..7 <- K = 8*khalf + 0..7 ; elems 8..15 <- K = 16 + 8*khalf + 0..7
static __device__ inline v16bf load_a(const __bf16* row, int khalf) {
    const __bf16* p = row + khalf * 8;
    return mkfrag(*(const v8bf*)p, *(const v8bf*)(p + 16));
}

// B-matrix fragment (32x16 bf16, KxN): lane = N col (lane&15), khalf = lane>>4.
// Same K order as A: elems 0..7 <- K = 8*khalf + 0..7 ; elems 8..15 <- K = 16 + 8*khalf + 0..7
static __device__ inline v16bf load_b(const __bf16* row, int khalf) {
    const __bf16* p = row + khalf * 8;
    return mkfrag(*(const v8bf*)p, *(const v8bf*)(p + 16));
}

static __device__ inline v8f wmma_bf16(v16bf a, v16bf b, v8f c) {
    // D = A x B + C   (16x16x32, f32 accumulate)
    return __builtin_amdgcn_wmma_f32_16x16x32_bf16(
        /*neg_a=*/false, a, /*neg_b=*/false, b,
        /*c_mod=*/(short)0, c, /*reuse_a=*/false, /*reuse_b=*/false);
}

// ---------------------------------------------------------------------------
// TDM: DMA a 2D bf16 tile (tile_h rows x tile_w elems, global row stride
// row_stride elems) into LDS at lds_addr, inserting 4 DWORDs of LDS
// padding after every row (row must be 32 DWORDs = 64 bf16 here).
// D# packing per CDNA5 ISA ch. 8 (group0 128b, group1 256b; groups 2/3 zero
// for 2D tensors).
// ---------------------------------------------------------------------------
#if __has_builtin(__builtin_amdgcn_tensor_load_to_lds)
#define USE_TDM 1
#else
#define USE_TDM 0
#endif

#if USE_TDM
static __device__ inline void tdm_load_tile_bf16(
    unsigned lds_addr, const void* gaddr,
    unsigned tile_w, unsigned tile_h, unsigned row_stride) {
    unsigned long long ga = (unsigned long long)(uintptr_t)gaddr;
    u32x4 g0;
    g0[0] = 1u;                                   // count=1 (valid), user mode
    g0[1] = lds_addr;                             // LDS byte address
    g0[2] = (unsigned)(ga & 0xFFFFFFFFu);         // global_addr[31:0]
    g0[3] = (unsigned)((ga >> 32) & 0x01FFFFFFu)  // global_addr[56:32]
            | (2u << 30);                         // type = 2 ("image")
    i32x8 g1;
    // data_size=1 (2B), pad_enable, pad_interval=4 (32 DW = one 64-elem row),
    // pad_amount=3 (4 DW = 16B = 8 bf16 of padding)
    g1[0] = (1 << 16) | (1 << 20) | (4 << 22) | (3 << 25);
    g1[1] = (int)(tile_w << 16);                  // tensor_dim0[15:0]
    g1[2] = (int)(tile_h << 16);                  // tensor_dim1[15:0]
    g1[3] = (int)(tile_w << 16);                  // tile_dim0
    g1[4] = (int)tile_h;                          // tile_dim1 (tile_dim2=0)
    g1[5] = (int)row_stride;                      // tensor_dim0_stride[31:0]
    g1[6] = 0;                                    // strides high bits
    g1[7] = 0;
    i32x4 gz = {0, 0, 0, 0};
#if defined(__clang_major__) && (__clang_major__ >= 23)
    i32x8 gz8 = {0, 0, 0, 0, 0, 0, 0, 0};
    __builtin_amdgcn_tensor_load_to_lds(g0, g1, gz, gz, gz8, 0);
#else
    __builtin_amdgcn_tensor_load_to_lds(g0, g1, gz, gz, 0);
#endif
}
#endif

// ---------------------------------------------------------------------------
// Projection GEMM: Y = relu(X @ W^T + bias)
//   X: [M, Kd]  (fp32 or bf16)   W: [N, Kd] fp32 (torch Linear weight)
//   Y: [M, N]   (bf16 workspace or fp32 final output)
// Block: 256 thr = 8 waves; block tile 128x128; wave tile 64x32; K-step 32.
// ---------------------------------------------------------------------------
template <bool IN_BF16, bool OUT_F32>
__global__ __launch_bounds__(256) void proj_kernel(
    const void* __restrict__ Xv, const float* __restrict__ W,
    const float* __restrict__ bias, void* __restrict__ Yv,
    int M, int N, int Kd) {
    constexpr int BM = 128, BN = 128, BK = 32, LDT = BK + 8;  // padded stride 40
    __shared__ __bf16 As[BM][LDT];
    __shared__ __bf16 Bs[BN][LDT];

    const int tid  = threadIdx.x;
    const int lane = tid & 31;
    const int wave = tid >> 5;
    const int wm   = wave >> 2;       // 0..1 -> M offset wm*64
    const int wn   = wave & 3;        // 0..3 -> N offset wn*32
    const int ln   = lane & 15;
    const int kh   = lane >> 4;
    const int m0   = blockIdx.y * BM;
    const int n0   = blockIdx.x * BN;

    const __bf16* Xb = (const __bf16*)Xv;
    const float*  Xf = (const float*)Xv;

    v8f acc[4][2];
#pragma unroll
    for (int i = 0; i < 4; ++i)
#pragma unroll
        for (int j = 0; j < 2; ++j) acc[i][j] = zero8();

    const int rowA = tid >> 1;           // 128 rows, 2 threads per row
    const int colA = (tid & 1) * 16;     // 16 contiguous K elems per thread

    for (int k0 = 0; k0 < Kd; k0 += BK) {
        __syncthreads();
        // ---- stage A tile (convert fp32 -> bf16 on the fly) ----
        if constexpr (IN_BF16) {
            const __bf16* src = Xb + (size_t)(m0 + rowA) * Kd + k0 + colA;
            *(v8bf*)&As[rowA][colA]     = *(const v8bf*)(src);
            *(v8bf*)&As[rowA][colA + 8] = *(const v8bf*)(src + 8);
        } else {
            const float* src = Xf + (size_t)(m0 + rowA) * Kd + k0 + colA;
#pragma unroll
            for (int i = 0; i < 4; ++i) {
                float4 f = *(const float4*)(src + i * 4);
                v4bf b4;
                b4[0] = (__bf16)f.x; b4[1] = (__bf16)f.y;
                b4[2] = (__bf16)f.z; b4[3] = (__bf16)f.w;
                *(v4bf*)&As[rowA][colA + i * 4] = b4;
            }
        }
        // ---- stage B tile (W rows = output features; always fp32) ----
        {
            const float* src = W + (size_t)(n0 + rowA) * Kd + k0 + colA;
#pragma unroll
            for (int i = 0; i < 4; ++i) {
                float4 f = *(const float4*)(src + i * 4);
                v4bf b4;
                b4[0] = (__bf16)f.x; b4[1] = (__bf16)f.y;
                b4[2] = (__bf16)f.z; b4[3] = (__bf16)f.w;
                *(v4bf*)&Bs[rowA][colA + i * 4] = b4;
            }
        }
        __syncthreads();

        v16bf af[4], bf2[2];
#pragma unroll
        for (int i = 0; i < 4; ++i)
            af[i] = load_a(&As[wm * 64 + i * 16 + ln][0], kh);
#pragma unroll
        for (int j = 0; j < 2; ++j)
            bf2[j] = load_b(&Bs[wn * 32 + j * 16 + ln][0], kh);
#pragma unroll
        for (int i = 0; i < 4; ++i)
#pragma unroll
            for (int j = 0; j < 2; ++j)
                acc[i][j] = wmma_bf16(af[i], bf2[j], acc[i][j]);
    }

    // ---- epilogue: bias + relu + store (C/D layout: n=lane&15, m=r+8*kh) ----
    float*  Yf = (float*)Yv;
    __bf16* Yb = (__bf16*)Yv;
#pragma unroll
    for (int j = 0; j < 2; ++j) {
        const int gn = n0 + wn * 32 + j * 16 + ln;
        const float bv = bias[gn];
#pragma unroll
        for (int i = 0; i < 4; ++i) {
#pragma unroll
            for (int r = 0; r < 8; ++r) {
                const int gm = m0 + wm * 64 + i * 16 + r + kh * 8;
                float v = acc[i][j][r] + bv;
                v = v > 0.0f ? v : 0.0f;
                if constexpr (OUT_F32) Yf[(size_t)gm * N + gn] = v;
                else                   Yb[(size_t)gm * N + gn] = (__bf16)v;
            }
        }
    }
}

// ---------------------------------------------------------------------------
// Flash attention: one block per (batch*head, 128-query tile); 8 waves,
// each wave owns 16 query rows. Key tiles of 32, causal.
// K tile staged into LDS by the Tensor Data Mover (wave 0 issues the D#,
// waits on TENSORcnt, workgroup barrier releases everyone); V tile staged
// transposed by regular loads so PV B-fragments are contiguous LDS rows.
// ---------------------------------------------------------------------------
__global__ __launch_bounds__(256) void attn_kernel(
    const __bf16* __restrict__ qh, const __bf16* __restrict__ kh,
    const __bf16* __restrict__ vh, __bf16* __restrict__ yo) {
    constexpr int S = 1024, D = 1024, HD = 64, KT = 32;

    __shared__ __bf16 Ks[32][72];      // [key][hd], padded (TDM pad feature)
    __shared__ __bf16 Vt[64][40];      // [hd][key], transposed, padded
    __shared__ __bf16 Ps[8][16][40];   // per-wave probs [q][key], padded

    const int tid  = threadIdx.x;
    const int lane = tid & 31;
    const int wave = tid >> 5;
    const int ln   = lane & 15;
    const int khf  = lane >> 4;

    const int bh  = blockIdx.y;
    const int b   = bh >> 4;            // H = 16
    const int h   = bh & 15;
    const int q0b = blockIdx.x * 128;
    const int q0w = q0b + wave * 16;

    // Q fragments (A layout), 2 K-steps covering HD=64
    const __bf16* qrow = qh + ((size_t)(b * S + q0w + ln)) * D + h * HD;
    v16bf qf[2];
    qf[0] = load_a(qrow, khf);
    qf[1] = load_a(qrow + 32, khf);

    v8f yacc[4];
#pragma unroll
    for (int j = 0; j < 4; ++j) yacc[j] = zero8();
    float mrow[8], lrow[8];
#pragma unroll
    for (int r = 0; r < 8; ++r) { mrow[r] = -1e30f; lrow[r] = 0.0f; }

    const int key  = tid >> 3;          // 32 keys, 8 threads/key
    const int hd   = (tid & 7) * 8;     // 8 bf16 per thread
    const int kend = q0b + 128;         // causal: later key tiles fully masked

#if USE_TDM
    const unsigned ks_lds = (unsigned)(uintptr_t)&Ks[0][0];
#endif

    for (int kt0 = 0; kt0 < kend; kt0 += KT) {
        __syncthreads();
        // ---- K tile: Tensor Data Mover global -> LDS (wave 0 issues) ----
#if USE_TDM
        if (wave == 0) {
            const __bf16* ksrc = kh + ((size_t)(b * S + kt0)) * D + h * HD;
            tdm_load_tile_bf16(ks_lds, ksrc, /*tile_w=*/HD, /*tile_h=*/KT,
                               /*row_stride=*/D);
        }
#else
        {
            const size_t rbase = ((size_t)(b * S + kt0 + key)) * D + h * HD + hd;
            *(v8bf*)&Ks[key][hd] = *(const v8bf*)(kh + rbase);
        }
#endif
        // ---- V tile: regular loads, transposed into LDS ----
        {
            const size_t rbase = ((size_t)(b * S + kt0 + key)) * D + h * HD + hd;
            v8bf vv = *(const v8bf*)(vh + rbase);
#pragma unroll
            for (int i = 0; i < 8; ++i) Vt[hd + i][key] = vv[i];  // transpose
        }
        if (kt0 + KT < kend) {  // prefetch next tile (global_prefetch_b8)
            const size_t nbase = ((size_t)(b * S + kt0 + KT + key)) * D + h * HD + hd;
            __builtin_prefetch(kh + nbase, 0, 1);
            __builtin_prefetch(vh + nbase, 0, 1);
        }
#if USE_TDM
        if (wave == 0) __builtin_amdgcn_s_wait_tensorcnt(0);
#endif
        __syncthreads();

        // ---- scores: S = Q K^T (two 16-key subtiles, two HD k-steps) ----
        v8f s0 = zero8(), s1 = zero8();
#pragma unroll
        for (int hs = 0; hs < 2; ++hs) {
            v16bf kb0 = load_b(&Ks[ln][hs * 32], khf);
            v16bf kb1 = load_b(&Ks[16 + ln][hs * 32], khf);
            s0 = wmma_bf16(qf[hs], kb0, s0);
            s1 = wmma_bf16(qf[hs], kb1, s1);
        }

        // ---- online softmax (rows live in 16-lane halves) ----
        float alpha[8];
#pragma unroll
        for (int r = 0; r < 8; ++r) {
            const int q   = q0w + r + khf * 8;
            const int k0i = kt0 + ln;
            const int k1i = kt0 + 16 + ln;
            float a = s0[r] * 0.125f;   // 1/sqrt(HD)
            float c = s1[r] * 0.125f;
            if (k0i > q) a = -1e30f;
            if (k1i > q) c = -1e30f;
            float t = fmaxf(a, c);
#pragma unroll
            for (int off = 1; off < 16; off <<= 1)
                t = fmaxf(t, __shfl_xor(t, off, 32));
            const float nm = fmaxf(mrow[r], t);
            alpha[r] = __expf(mrow[r] - nm);
            mrow[r]  = nm;
            const float e0 = __expf(a - nm);
            const float e1 = __expf(c - nm);
            float rs = e0 + e1;
#pragma unroll
            for (int off = 1; off < 16; off <<= 1)
                rs += __shfl_xor(rs, off, 32);
            lrow[r] = lrow[r] * alpha[r] + rs;
            const int m = r + khf * 8;                 // C-layout row
            Ps[wave][m][ln]      = (__bf16)e0;         // re-layout via LDS
            Ps[wave][m][16 + ln] = (__bf16)e1;
        }
#pragma unroll
        for (int j = 0; j < 4; ++j)
#pragma unroll
            for (int r = 0; r < 8; ++r) yacc[j][r] *= alpha[r];

        // ---- PV: Y += P V (A from wave-private LDS, B = V^T rows) ----
        v16bf pa = load_a(&Ps[wave][ln][0], khf);
#pragma unroll
        for (int j = 0; j < 4; ++j) {
            v16bf vb = load_b(&Vt[j * 16 + ln][0], khf);
            yacc[j] = wmma_bf16(pa, vb, yacc[j]);
        }
    }

    // ---- finalize: divide by row sums, store merged-head bf16 ----
#pragma unroll
    for (int r = 0; r < 8; ++r) lrow[r] = 1.0f / lrow[r];
#pragma unroll
    for (int j = 0; j < 4; ++j) {
#pragma unroll
        for (int r = 0; r < 8; ++r) {
            const int row = q0w + r + khf * 8;
            const int hdo = j * 16 + ln;
            yo[((size_t)(b * S + row)) * D + h * HD + hdo] =
                (__bf16)(yacc[j][r] * lrow[r]);
        }
    }
}

// ---------------------------------------------------------------------------
extern "C" void kernel_launch(void* const* d_in, const int* in_sizes, int n_in,
                              void* d_out, int out_size, void* d_ws, size_t ws_size,
                              hipStream_t stream) {
    (void)in_sizes; (void)n_in; (void)out_size; (void)ws_size;
    constexpr int B = 4, S = 1024, D = 1024;
    constexpr size_t MS = (size_t)B * S * D;  // elements per [B,S,D] tensor

    const float* q  = (const float*)d_in[0];
    const float* k  = (const float*)d_in[1];
    const float* v  = (const float*)d_in[2];
    // d_in[3] = mask: exactly causal tril, applied analytically in attn_kernel
    const float* Wq = (const float*)d_in[4];  const float* bq = (const float*)d_in[5];
    const float* Wk = (const float*)d_in[6];  const float* bk = (const float*)d_in[7];
    const float* Wv = (const float*)d_in[8];  const float* bv = (const float*)d_in[9];
    const float* Wo = (const float*)d_in[10]; const float* bo = (const float*)d_in[11];
    float* out = (float*)d_out;

    __bf16* qh  = (__bf16*)d_ws;   // 8 MB each
    __bf16* khp = qh + MS;
    __bf16* vhp = khp + MS;
    __bf16* yb  = vhp + MS;

    dim3 pgrid(D / 128, (B * S) / 128);   // (8, 32)
    dim3 blk(256);

    proj_kernel<false, false><<<pgrid, blk, 0, stream>>>(q, Wq, bq, qh,  B * S, D, D);
    proj_kernel<false, false><<<pgrid, blk, 0, stream>>>(k, Wk, bk, khp, B * S, D, D);
    proj_kernel<false, false><<<pgrid, blk, 0, stream>>>(v, Wv, bv, vhp, B * S, D, D);

    dim3 agrid(S / 128, B * 16);          // (8, 64)
    attn_kernel<<<agrid, blk, 0, stream>>>(qh, khp, vhp, yb);

    proj_kernel<true, true><<<pgrid, blk, 0, stream>>>(yb, Wo, bo, out, B * S, D, D);
}
